// TriangleAttention_76785425318055
// MI455X (gfx1250) — matrix operand order repaired and checked by reading, not compiled, not run on hardware
//
#include <hip/hip_runtime.h>
#include <hip/hip_bf16.h>
#include <math.h>

typedef _Float16 h4  __attribute__((ext_vector_type(4)));
typedef _Float16 h8  __attribute__((ext_vector_type(8)));
typedef _Float16 h16 __attribute__((ext_vector_type(16)));
typedef float    f8  __attribute__((ext_vector_type(8)));
typedef unsigned int u32x4 __attribute__((ext_vector_type(4)));
typedef int          i32x4 __attribute__((ext_vector_type(4)));
typedef int          i32x8 __attribute__((ext_vector_type(8)));

#define WMMA_F16(a, b, c) \
  __builtin_amdgcn_wmma_f32_16x16x32_f16(false, (a), false, (b), (short)0, (c), false, false)
#define CAT8_8(lo, hi) \
  __builtin_shufflevector((lo), (hi), 0,1,2,3,4,5,6,7,8,9,10,11,12,13,14,15)

#ifndef __has_builtin
#define __has_builtin(x) 0
#endif
#if __has_builtin(__builtin_amdgcn_tensor_load_to_lds)
#define USE_TDM 1
#else
#define USE_TDM 0
#endif

constexpr int I_DIM = 256;
constexpr int J_DIM = 256;
constexpr int C_DIM = 128;
constexpr int H_NUM = 4;
constexpr int CH    = 32;
constexpr int ROWS  = I_DIM * J_DIM;   // 65536

// ---------------------------------------------------------------------------
// 1) LayerNorm: one wave32 per row of 128 floats, writes f16 xn
// ---------------------------------------------------------------------------
__global__ void ln_kernel(const float* __restrict__ x,
                          const float* __restrict__ w,
                          const float* __restrict__ b,
                          _Float16* __restrict__ xn) {
  int wave = threadIdx.x >> 5;
  int lane = threadIdx.x & 31;
  int row  = blockIdx.x * 8 + wave;
  const float* xr = x + (size_t)row * C_DIM;
  float4 v = *(const float4*)(xr + lane * 4);
  float s1 = v.x + v.y + v.z + v.w;
  float s2 = v.x * v.x + v.y * v.y + v.z * v.z + v.w * v.w;
#pragma unroll
  for (int m = 16; m >= 1; m >>= 1) {
    s1 += __shfl_xor(s1, m, 32);
    s2 += __shfl_xor(s2, m, 32);
  }
  float mu  = s1 * (1.0f / 128.0f);
  float var = s2 * (1.0f / 128.0f) - mu * mu;
  float inv = rsqrtf(var + 1e-5f);
  float4 wv = *(const float4*)(w + lane * 4);
  float4 bv = *(const float4*)(b + lane * 4);
  h4 o;
  o[0] = (_Float16)((v.x - mu) * inv * wv.x + bv.x);
  o[1] = (_Float16)((v.y - mu) * inv * wv.y + bv.y);
  o[2] = (_Float16)((v.z - mu) * inv * wv.z + bv.z);
  o[3] = (_Float16)((v.w - mu) * inv * wv.w + bv.w);
  *(h4*)(xn + (size_t)row * C_DIM + lane * 4) = o;
}

// ---------------------------------------------------------------------------
// 2) Weight prep: transpose 128x128 f32 weights -> f16 Wt[n][k] (K-contiguous,
//    matching the WMMA B-operand layout: lane = column n, K striped in VGPRs)
// ---------------------------------------------------------------------------
__global__ void wprep_kernel(const float* __restrict__ wq,
                             const float* __restrict__ wk,
                             const float* __restrict__ wv,
                             const float* __restrict__ wg,
                             const float* __restrict__ wo,
                             _Float16* __restrict__ wt) {
  int m = blockIdx.y;
  const float* W = (m == 0) ? wq : (m == 1) ? wk : (m == 2) ? wv : (m == 3) ? wg : wo;
  int idx = blockIdx.x * 256 + threadIdx.x;  // 0..16383
  int n = idx >> 7, k = idx & 127;
  wt[(size_t)m * 16384 + (size_t)n * 128 + k] = (_Float16)W[(size_t)k * 128 + n];
}

// ---------------------------------------------------------------------------
// 3) Triangle bias: tri[h*65536 + row] = xn[row,:] . w_tri[:,h]   (N=4 -> VALU)
// ---------------------------------------------------------------------------
__global__ void tri_kernel(const _Float16* __restrict__ xn,
                           const float* __restrict__ w_tri,
                           float* __restrict__ tri) {
  int row = blockIdx.x * 256 + threadIdx.x;
  const _Float16* xr = xn + (size_t)row * C_DIM;
  float acc[4] = {0.f, 0.f, 0.f, 0.f};
#pragma unroll 4
  for (int c = 0; c < C_DIM; ++c) {
    float xv = (float)xr[c];
#pragma unroll
    for (int h = 0; h < H_NUM; ++h) acc[h] += xv * w_tri[c * 4 + h];
  }
#pragma unroll
  for (int h = 0; h < H_NUM; ++h) tri[(size_t)h * ROWS + row] = acc[h];
}

// ---------------------------------------------------------------------------
// 4) 128xK128xN128 WMMA GEMM; A tile staged to LDS via TDM (tensor_load_to_lds)
//    out = (A @ Bt^T) * scale + bias ; store f16 (outh) or f32 (outf)
// ---------------------------------------------------------------------------
__global__ void gemm128_kernel(const _Float16* __restrict__ A,
                               const _Float16* __restrict__ Bt,
                               const float* __restrict__ bias,
                               float scale,
                               _Float16* __restrict__ outh,
                               float* __restrict__ outf) {
  __shared__ _Float16 At[128 * 128];  // 32 KB
  const int tid = threadIdx.x, lane = tid & 31, wave = tid >> 5;
  const int lane16 = lane & 15, hi = (lane >= 16) ? 1 : 0;
  const int rowBase = blockIdx.x * 128;

  // LDS byte offset of At (low 32 bits of the generic/aperture address).
  // The ptrtoint both feeds the TDM descriptor and escapes At's address so
  // the asm memory clobber below makes the TDM's LDS writes visible.
  unsigned lds_off = (unsigned)(unsigned long long)(_Float16*)At;

#if USE_TDM
  if (wave == 0) {
    unsigned long long ga = (unsigned long long)(A + (size_t)rowBase * 128);
    u32x4 g0;
    g0[0] = 1u;                                   // count=1 (valid descriptor)
    g0[1] = lds_off;                              // lds_addr
    g0[2] = (unsigned)ga;                         // global_addr[31:0]
    g0[3] = (unsigned)((ga >> 32) & 0x1FFFFFFull) // global_addr[56:32]
            | (2u << 30);                         // type=2 ("image")
    i32x8 g1;
    g1[0] = 0x00010000;        // workgroup_mask=0, data_size=1 (2 bytes)
    g1[1] = (128 << 16);       // tensor_dim0 lo16 = 128
    g1[2] = 0;                 // tensor_dim0 hi / tensor_dim1 lo16 = 0 (65536)
    g1[3] = (128 << 16) | 1;   // tile_dim0=128 | tensor_dim1 hi16 = 1
    g1[4] = 128;               // tile_dim1=128, tile_dim2=0
    g1[5] = 128;               // tensor_dim0_stride = 128 elements
    g1[6] = 0;
    g1[7] = 0;
    i32x4 gz4 = {0, 0, 0, 0};
    i32x8 gz8 = {0, 0, 0, 0, 0, 0, 0, 0};
    __builtin_amdgcn_tensor_load_to_lds(g0, g1, gz4, gz4, gz8, 0);
    __builtin_amdgcn_s_wait_tensorcnt(0);
  }
  __syncthreads();
  // At's address escaped via lds_off; this opaque asm may have written it,
  // so the At reads below must be real ds_read ops (not folded to undef).
  asm volatile("" : : "s"(lds_off) : "memory");
#else
  {
    const uint4* src = (const uint4*)(A + (size_t)rowBase * 128);
    uint4* dst = (uint4*)At;
#pragma unroll
    for (int it = 0; it < 8; ++it) dst[it * 256 + tid] = src[it * 256 + tid];
  }
  __syncthreads();
  (void)lds_off;
#endif

  f8 acc[8] = {};
  const int arow = wave * 16 + lane16;
#pragma unroll
  for (int ks = 0; ks < 4; ++ks) {
    // A fragment: lane<16 holds K {0..15}, lane>=16 holds {16..31}
    int ka = ks * 32 + (hi ? 16 : 0);
    h8 a0 = *(const h8*)&At[arow * 128 + ka];
    h8 a1 = *(const h8*)&At[arow * 128 + ka + 8];
    h16 af = CAT8_8(a0, a1);
    int kb = ks * 32 + (hi ? 16 : 0);  // B: half-wave K split, contiguous 16
#pragma unroll
    for (int nt = 0; nt < 8; ++nt) {
      int col = nt * 16 + lane16;
      h16 bf = *(const h16*)&Bt[(size_t)col * 128 + kb];
      acc[nt] = WMMA_F16(af, bf, acc[nt]);
    }
  }

  const int rowOut = rowBase + wave * 16;
#pragma unroll
  for (int nt = 0; nt < 8; ++nt) {
    int n = nt * 16 + lane16;
    float bb = bias ? bias[n] : 0.0f;
#pragma unroll
    for (int r = 0; r < 8; ++r) {
      int m = rowOut + r + hi * 8;   // C/D layout: VGPR r -> row r (+8 hi half)
      float v = acc[nt][r] * scale + bb;
      if (outf) outf[(size_t)m * 128 + n] = v;
      else      outh[(size_t)m * 128 + n] = (_Float16)v;
    }
  }
}

// ---------------------------------------------------------------------------
// 5) Attention per (i,h): S=QK^T(+mask,+tri) -> softmax -> O=AV -> sigmoid gate
// ---------------------------------------------------------------------------
__global__ void attn_kernel(const _Float16* __restrict__ Q,
                            const _Float16* __restrict__ K,
                            const _Float16* __restrict__ V,
                            const _Float16* __restrict__ G,
                            const float* __restrict__ mask,
                            const float* __restrict__ tri,
                            _Float16* __restrict__ O) {
  extern __shared__ char smem[];
  _Float16* Kt    = (_Float16*)smem;             // [256][32]  row-major (B op)
  _Float16* Vt    = Kt + 256 * 32;               // [32][256]  transposed (B op)
  float*    maskb = (float*)(Vt + 32 * 256);     // [256]
  _Float16* Al    = (_Float16*)(maskb + 256);    // [8 waves][16][256]

  const int i = blockIdx.x, h = blockIdx.y;
  const int tid = threadIdx.x, lane = tid & 31, wave = tid >> 5;
  const int lane16 = lane & 15, hi = (lane >= 16) ? 1 : 0;

  {  // stage K (row-major), V (transposed), mask bias
    int j = tid;
    const _Float16* kr = K + ((size_t)(i * 256 + j) * 128 + h * 32);
    const uint4* ks = (const uint4*)kr;
    uint4* kd = (uint4*)&Kt[j * 32];
    kd[0] = ks[0]; kd[1] = ks[1]; kd[2] = ks[2]; kd[3] = ks[3];
    const _Float16* vr = V + ((size_t)(i * 256 + j) * 128 + h * 32);
#pragma unroll
    for (int c = 0; c < 32; ++c) Vt[c * 256 + j] = vr[c];
    maskb[j] = 1.0e9f * (mask[(size_t)i * 256 + j] - 1.0f);
  }
  __syncthreads();

  _Float16* Aw = Al + wave * 16 * 256;
  const float* trih = tri + (size_t)h * ROWS;

  for (int qt = 0; qt < 2; ++qt) {
    const int qbase = (wave * 2 + qt) * 16;
    // Q fragment (Q already scaled by CH^-0.5)
    const _Float16* qr = Q + ((size_t)(i * 256 + qbase + lane16) * 128 + h * 32);
    int ka = hi ? 16 : 0;
    h8 q0 = *(const h8*)(qr + ka);
    h8 q1 = *(const h8*)(qr + ka + 8);
    h16 qf = CAT8_8(q0, q1);

    f8 acc[16] = {};
    int kb = hi ? 16 : 0;
#pragma unroll
    for (int nt = 0; nt < 16; ++nt) {
      int col = nt * 16 + lane16;
      h16 bf = *(const h16*)&Kt[col * 32 + kb];
      acc[nt] = WMMA_F16(qf, bf, acc[nt]);
    }

    // add mask + triangle bias
#pragma unroll
    for (int nt = 0; nt < 16; ++nt) {
      int n = nt * 16 + lane16;
      float mb = maskb[n];
#pragma unroll
      for (int r = 0; r < 8; ++r) {
        int q = qbase + r + hi * 8;
        acc[nt][r] += mb + trih[(size_t)q * 256 + n];
      }
    }

    // row softmax (row lives in one VGPR across lane%16 of 16 n-tiles)
    float mx[8], sm[8];
#pragma unroll
    for (int r = 0; r < 8; ++r) mx[r] = -3.0e38f;
#pragma unroll
    for (int nt = 0; nt < 16; ++nt)
#pragma unroll
      for (int r = 0; r < 8; ++r) mx[r] = fmaxf(mx[r], acc[nt][r]);
#pragma unroll
    for (int r = 0; r < 8; ++r)
#pragma unroll
      for (int m = 8; m >= 1; m >>= 1) mx[r] = fmaxf(mx[r], __shfl_xor(mx[r], m, 32));
#pragma unroll
    for (int r = 0; r < 8; ++r) sm[r] = 0.0f;
#pragma unroll
    for (int nt = 0; nt < 16; ++nt)
#pragma unroll
      for (int r = 0; r < 8; ++r) {
        float p = __expf(acc[nt][r] - mx[r]);
        acc[nt][r] = p;
        sm[r] += p;
      }
#pragma unroll
    for (int r = 0; r < 8; ++r)
#pragma unroll
      for (int m = 8; m >= 1; m >>= 1) sm[r] += __shfl_xor(sm[r], m, 32);
    float rcp[8];
#pragma unroll
    for (int r = 0; r < 8; ++r) rcp[r] = 1.0f / sm[r];

    // stash normalized probs (f16) as A-operand tile for second GEMM
#pragma unroll
    for (int nt = 0; nt < 16; ++nt)
#pragma unroll
      for (int r = 0; r < 8; ++r) {
        int mrow = r + hi * 8;
        Aw[mrow * 256 + nt * 16 + lane16] = (_Float16)(acc[nt][r] * rcp[r]);
      }
    asm volatile("s_wait_dscnt 0" ::: "memory");  // wave-local LDS RAW

    // O = A[16x256] @ V[256x32]
    f8 oa[2] = {};
#pragma unroll
    for (int ks = 0; ks < 8; ++ks) {
      int ca = ks * 32 + (hi ? 16 : 0);
      h8 p0 = *(const h8*)&Aw[lane16 * 256 + ca];
      h8 p1 = *(const h8*)&Aw[lane16 * 256 + ca + 8];
      h16 af = CAT8_8(p0, p1);
      int cv = ks * 32 + (hi ? 16 : 0);
#pragma unroll
      for (int nt = 0; nt < 2; ++nt) {
        int col = nt * 16 + lane16;
        h16 bf = *(const h16*)&Vt[col * 256 + cv];
        oa[nt] = WMMA_F16(af, bf, oa[nt]);
      }
    }

    // sigmoid gate and store f16
#pragma unroll
    for (int nt = 0; nt < 2; ++nt) {
      int ch = nt * 16 + lane16;
#pragma unroll
      for (int r = 0; r < 8; ++r) {
        int q = qbase + r + hi * 8;
        size_t idx = (size_t)(i * 256 + q) * 128 + h * 32 + ch;
        float gv = (float)G[idx];
        float gs = 1.0f / (1.0f + __expf(-gv));
        O[idx] = (_Float16)(oa[nt][r] * gs);
      }
    }
  }
}

// ---------------------------------------------------------------------------
// launch
// ---------------------------------------------------------------------------
extern "C" void kernel_launch(void* const* d_in, const int* in_sizes, int n_in,
                              void* d_out, int out_size, void* d_ws, size_t ws_size,
                              hipStream_t stream) {
  const float* x     = (const float*)d_in[0];
  const float* mask  = (const float*)d_in[1];
  const float* ln_w  = (const float*)d_in[3];
  const float* ln_b  = (const float*)d_in[4];
  const float* w_tri = (const float*)d_in[5];
  const float* wq    = (const float*)d_in[6];
  const float* wk    = (const float*)d_in[7];
  const float* wv    = (const float*)d_in[8];
  const float* wg    = (const float*)d_in[9];
  const float* bg    = (const float*)d_in[10];
  const float* wo    = (const float*)d_in[11];
  const float* bo    = (const float*)d_in[12];
  float* out = (float*)d_out;

  char* ws = (char*)d_ws;
  const size_t mat_h = (size_t)ROWS * C_DIM * sizeof(_Float16);  // 16 MB
  _Float16* xn   = (_Float16*)ws; ws += mat_h;
  _Float16* qbuf = (_Float16*)ws; ws += mat_h;
  _Float16* kbuf = (_Float16*)ws; ws += mat_h;
  _Float16* vbuf = (_Float16*)ws; ws += mat_h;
  _Float16* gbuf = (_Float16*)ws; ws += mat_h;
  _Float16* obuf = (_Float16*)ws; ws += mat_h;
  float*    tri  = (float*)ws;    ws += (size_t)H_NUM * ROWS * sizeof(float);
  _Float16* wt   = (_Float16*)ws; ws += (size_t)5 * 16384 * sizeof(_Float16);

  ln_kernel<<<ROWS / 8, 256, 0, stream>>>(x, ln_w, ln_b, xn);
  wprep_kernel<<<dim3(64, 5), 256, 0, stream>>>(wq, wk, wv, wg, wo, wt);
  tri_kernel<<<ROWS / 256, 256, 0, stream>>>(xn, w_tri, tri);

  const float qscale = 0.17677669529663687f;  // 32^-0.5
  gemm128_kernel<<<512, 256, 0, stream>>>(xn, wt + 0 * 16384, nullptr, qscale, qbuf, nullptr);
  gemm128_kernel<<<512, 256, 0, stream>>>(xn, wt + 1 * 16384, nullptr, 1.0f,   kbuf, nullptr);
  gemm128_kernel<<<512, 256, 0, stream>>>(xn, wt + 2 * 16384, nullptr, 1.0f,   vbuf, nullptr);
  gemm128_kernel<<<512, 256, 0, stream>>>(xn, wt + 3 * 16384, bg,      1.0f,   gbuf, nullptr);

  const size_t smem = (256 * 32 + 32 * 256 + 8 * 16 * 256) * sizeof(_Float16)
                    + 256 * sizeof(float);    // ~97 KB < 320 KB/WGP
  attn_kernel<<<dim3(I_DIM, H_NUM), 256, smem, stream>>>(qbuf, kbuf, vbuf, gbuf,
                                                         mask, tri, obuf);

  gemm128_kernel<<<512, 256, 0, stream>>>(obuf, wt + 4 * 16384, bo, 1.0f, nullptr, out);
}
